// MaskedMultiHeadAttention_21406117003304
// MI455X (gfx1250) — matrix operand order repaired and checked by reading, not compiled, not run on hardware
//
#include <hip/hip_runtime.h>

typedef __attribute__((ext_vector_type(16))) __bf16 v16bf;
typedef __attribute__((ext_vector_type(8)))  float  v8f;

#define N_HEADS  16
#define HEAD_DIM 64
#define SRC_LEN  64
#define TSEQ     2048
#define BB       4
#define EDIM     1024
#define MROWS    (BB * TSEQ)

union Frag { v16bf v; uint4 u[2]; };

__device__ __forceinline__ unsigned short f2bf(float f) {
    unsigned int u = __float_as_uint(f);
    u += 0x7FFFu + ((u >> 16) & 1u);
    return (unsigned short)(u >> 16);
}

__device__ __forceinline__ v8f bfwmma(const Frag& a, const Frag& b, v8f c) {
    return __builtin_amdgcn_wmma_f32_16x16x32_bf16(false, a.v, false, b.v,
                                                   (short)0, c, false, false);
}

// ---------------- conversion kernels ----------------

__global__ void cvt_x_kernel(const float* __restrict__ in,
                             unsigned short* __restrict__ out, int n) {
    int i = blockIdx.x * blockDim.x + threadIdx.x;
    if (i < n) out[i] = f2bf(in[i]);
}

// Wt[n][k] = W[k][n]  (bf16)
__global__ void cvt_wt_kernel(const float* __restrict__ W,
                              unsigned short* __restrict__ Wt) {
    int i = blockIdx.x * blockDim.x + threadIdx.x;
    int n = i >> 10, k = i & 1023;
    Wt[i] = f2bf(W[k * EDIM + n]);
}

// ---------------- 32x64 register-blocked GEMM mainloop (K = EDIM) ----------------
// acc[mt][ng]: mt in {0,1} row subtiles of 16, ng in {0..3} col subtiles of 16.
// Per k-step: 4 A loads + 8 B loads feed 8 WMMAs (1.5 b128 loads / WMMA).

__device__ __forceinline__ void gemm32x64(const unsigned short* __restrict__ A,
                                          const unsigned short* __restrict__ Bt,
                                          int mbase, int nbase, int l16, int half,
                                          v8f acc[2][4]) {
    const unsigned short* ap0 = A + (size_t)(mbase + l16) * EDIM;
    const unsigned short* ap1 = ap0 + (size_t)16 * EDIM;
    const unsigned short* bp0 = Bt + (size_t)(nbase + l16) * EDIM;
    const int a0 = half * 16;
    const int b0 = half * 16;
#pragma unroll
    for (int mt = 0; mt < 2; ++mt)
#pragma unroll
        for (int ng = 0; ng < 4; ++ng) acc[mt][ng] = {};

#pragma unroll 2
    for (int k0 = 0; k0 < EDIM; k0 += 32) {
        Frag fa0, fa1;
        fa0.u[0] = *reinterpret_cast<const uint4*>(ap0 + k0 + a0);
        fa0.u[1] = *reinterpret_cast<const uint4*>(ap0 + k0 + a0 + 8);
        fa1.u[0] = *reinterpret_cast<const uint4*>(ap1 + k0 + a0);
        fa1.u[1] = *reinterpret_cast<const uint4*>(ap1 + k0 + a0 + 8);
#pragma unroll
        for (int ng = 0; ng < 4; ++ng) {
            const unsigned short* bp = bp0 + (size_t)(ng * 16) * EDIM + k0 + b0;
            Frag fb;
            fb.u[0] = *reinterpret_cast<const uint4*>(bp);
            fb.u[1] = *reinterpret_cast<const uint4*>(bp + 8);
            acc[0][ng] = bfwmma(fa0, fb, acc[0][ng]);
            acc[1][ng] = bfwmma(fa1, fb, acc[1][ng]);
        }
    }
}

// ---------------- QKV projection + bias + RoPE + layout transform ----------------
// z=0: Q -> Qr[B,H,T,D] (rope, *0.125)   z=1: K -> Kr[B,H,T,D] (rope)
// z=2: V -> Vt[B,H,D,T]
// One wave: 32 token rows x one full head (64 cols).

__global__ void __launch_bounds__(256)
qkv_rope_kernel(const unsigned short* __restrict__ xb,
                const unsigned short* __restrict__ Wtq,
                const unsigned short* __restrict__ Wtk,
                const unsigned short* __restrict__ Wtv,
                const float* __restrict__ bq,
                const float* __restrict__ bk,
                const float* __restrict__ bv,
                unsigned short* __restrict__ Qr,
                unsigned short* __restrict__ Kr,
                unsigned short* __restrict__ Vt) {
    const int lane  = threadIdx.x & 31;
    const int wave  = threadIdx.x >> 5;
    const int half  = lane >> 4;
    const int l16   = lane & 15;
    const int z     = blockIdx.z;
    const int mbase = blockIdx.x * 32;
    const int nbase = (blockIdx.y * 8 + wave) * 64;   // one head per wave

    const unsigned short* Wt = (z == 0) ? Wtq : (z == 1) ? Wtk : Wtv;
    const float* bias        = (z == 0) ? bq  : (z == 1) ? bk  : bv;

    v8f acc[2][4];
    gemm32x64(xb, Wt, mbase, nbase, l16, half, acc);

    const int b  = mbase / TSEQ;          // 32-row tile never straddles a batch
    const int t0 = mbase % TSEQ;
    const int h  = nbase / HEAD_DIM;
    const int bh = b * N_HEADS + h;

    if (z == 2) {
#pragma unroll
        for (int ng = 0; ng < 4; ++ng) {
            const int d = ng * 16 + l16;
            const float bsc = bias[nbase + d];
            unsigned short* vrow = Vt + (size_t)(bh * HEAD_DIM + d) * TSEQ + t0;
#pragma unroll
            for (int mt = 0; mt < 2; ++mt)
#pragma unroll
                for (int r = 0; r < 8; ++r)
                    vrow[mt * 16 + r + half * 8] = f2bf(acc[mt][ng][r] + bsc);
        }
    } else {
        unsigned short* out = (z == 0) ? Qr : Kr;
        const float scale = (z == 0) ? 0.125f : 1.0f;  // HEAD_DIM^-0.5 folded into Q
#pragma unroll
        for (int ng = 0; ng < 4; ++ng) {
            const int d = ng * 16 + l16;
            const float bsc  = bias[nbase + d];
            const float invf = __expf(-(float)(d & ~1) * (9.210340371976184f / 64.0f));
#pragma unroll
            for (int mt = 0; mt < 2; ++mt)
#pragma unroll
            for (int r = 0; r < 8; ++r) {
                float v  = acc[mt][ng][r] + bsc;
                float pv = __shfl_xor(v, 1, 32);       // rope partner (d^1 column)
                int   t  = t0 + mt * 16 + r + half * 8;
                float s, c;
                __sincosf((float)t * invf, &s, &c);
                float o = (d & 1) ? (pv * s + v * c) : (v * c - pv * s);
                out[(size_t)(bh * TSEQ + t) * HEAD_DIM + d] = f2bf(o * scale);
            }
        }
    }
}

// ---------------- flash attention (one wave = one 16-row q tile) ----------------

__global__ void __launch_bounds__(128)
attn_kernel(const unsigned short* __restrict__ Qr,
            const unsigned short* __restrict__ Kr,
            const unsigned short* __restrict__ Vt,
            unsigned short* __restrict__ Oa) {
    __shared__ __align__(16) unsigned short pbuf[4][16 * 32];
    const int lane  = threadIdx.x & 31;
    const int wave  = threadIdx.x >> 5;
    const int half  = lane >> 4;
    const int half8 = half * 8;
    const int l16   = lane & 15;
    const int bh    = blockIdx.y;
    const int qbase = (blockIdx.x * 4 + wave) * 16;
    unsigned short* pb = pbuf[wave];

    // Q A-fragments (two 32-wide d chunks), scaled by 1/8 already
    Frag aq[2];
    const unsigned short* qp = Qr + (size_t)(bh * TSEQ + qbase + l16) * HEAD_DIM;
#pragma unroll
    for (int c = 0; c < 2; ++c) {
        aq[c].u[0] = *reinterpret_cast<const uint4*>(qp + c * 32 + half * 16);
        aq[c].u[1] = *reinterpret_cast<const uint4*>(qp + c * 32 + half * 16 + 8);
    }

    v8f o0 = {}, o1 = {}, o2 = {}, o3 = {};
    float mrow[8], lrow[8];
#pragma unroll
    for (int r = 0; r < 8; ++r) { mrow[r] = -3.0e38f; lrow[r] = 0.0f; }

    const int jend = min(TSEQ, qbase + 16 + SRC_LEN);   // allowed: j <= i + 64
    const unsigned short* kbase = Kr + (size_t)bh * TSEQ * HEAD_DIM;
    const unsigned short* vbase = Vt + (size_t)bh * HEAD_DIM * TSEQ;

    for (int jb = 0; jb < jend; jb += 32) {
        // ---- S = Q K^T for 32 keys (two 16-key subtiles) ----
        v8f s0 = {}, s1 = {};
#pragma unroll
        for (int c = 0; c < 2; ++c) {
            const unsigned short* kp0 = kbase + (size_t)(jb + l16) * HEAD_DIM;
            const unsigned short* kp1 = kbase + (size_t)(jb + 16 + l16) * HEAD_DIM;
            Frag bk0, bk1;
            bk0.u[0] = *reinterpret_cast<const uint4*>(kp0 + c * 32 + half * 16);
            bk0.u[1] = *reinterpret_cast<const uint4*>(kp0 + c * 32 + half * 16 + 8);
            bk1.u[0] = *reinterpret_cast<const uint4*>(kp1 + c * 32 + half * 16);
            bk1.u[1] = *reinterpret_cast<const uint4*>(kp1 + c * 32 + half * 16 + 8);
            s0 = bfwmma(aq[c], bk0, s0);
            s1 = bfwmma(aq[c], bk1, s1);
        }

        // ---- mask + online softmax ----
        const int j0 = jb + l16, j1 = jb + 16 + l16;
        float rmax[8];
#pragma unroll
        for (int r = 0; r < 8; ++r) {
            int i_row = qbase + r + half8;
            float v0 = (j0 <= i_row + SRC_LEN) ? s0[r] : -3.0e38f;
            float v1 = (j1 <= i_row + SRC_LEN) ? s1[r] : -3.0e38f;
            s0[r] = v0; s1[r] = v1;
            rmax[r] = fmaxf(v0, v1);
        }
#pragma unroll
        for (int msk = 1; msk <= 8; msk <<= 1)
#pragma unroll
            for (int r = 0; r < 8; ++r)
                rmax[r] = fmaxf(rmax[r], __shfl_xor(rmax[r], msk, 32));

        float alpha[8], rsum[8];
#pragma unroll
        for (int r = 0; r < 8; ++r) {
            float mn = fmaxf(mrow[r], rmax[r]);
            alpha[r] = __expf(mrow[r] - mn);
            mrow[r]  = mn;
            float p0 = __expf(s0[r] - mn);
            float p1 = __expf(s1[r] - mn);
            s0[r] = p0; s1[r] = p1;
            rsum[r] = p0 + p1;
        }
#pragma unroll
        for (int msk = 1; msk <= 8; msk <<= 1)
#pragma unroll
            for (int r = 0; r < 8; ++r)
                rsum[r] += __shfl_xor(rsum[r], msk, 32);
#pragma unroll
        for (int r = 0; r < 8; ++r) {
            lrow[r] = lrow[r] * alpha[r] + rsum[r];
            o0[r] *= alpha[r]; o1[r] *= alpha[r];
            o2[r] *= alpha[r]; o3[r] *= alpha[r];
        }

        // ---- stage P (bf16) through per-wave LDS, re-read in A-fragment order ----
#pragma unroll
        for (int r = 0; r < 8; ++r) {
            int row = r + half8;
            pb[row * 32 + l16]      = f2bf(s0[r]);
            pb[row * 32 + 16 + l16] = f2bf(s1[r]);
        }
        asm volatile("s_wait_dscnt 0x0" ::: "memory");
        Frag ap;
        ap.u[0] = *reinterpret_cast<const uint4*>(&pb[l16 * 32 + half * 16]);
        ap.u[1] = *reinterpret_cast<const uint4*>(&pb[l16 * 32 + half * 16 + 8]);

        // ---- O += P V  (4 d-subtiles of 16) ----
#pragma unroll
        for (int dt = 0; dt < 4; ++dt) {
            const unsigned short* vrow =
                vbase + (size_t)(dt * 16 + l16) * TSEQ + jb + half * 16;
            Frag bv;
            bv.u[0] = *reinterpret_cast<const uint4*>(vrow);
            bv.u[1] = *reinterpret_cast<const uint4*>(vrow + 8);
            v8f& o = (dt == 0) ? o0 : (dt == 1) ? o1 : (dt == 2) ? o2 : o3;
            o = bfwmma(ap, bv, o);
        }
    }

    // ---- finalize: O /= l, write bf16 [B*T, E] for the final GEMM ----
    const int b = bh >> 4, h = bh & 15;
#pragma unroll
    for (int r = 0; r < 8; ++r) {
        float inv = 1.0f / lrow[r];
        int q = qbase + r + half8;
        size_t base = (size_t)(b * TSEQ + q) * EDIM + h * HEAD_DIM + l16;
        Oa[base + 0]  = f2bf(o0[r] * inv);
        Oa[base + 16] = f2bf(o1[r] * inv);
        Oa[base + 32] = f2bf(o2[r] * inv);
        Oa[base + 48] = f2bf(o3[r] * inv);
    }
}

// ---------------- final projection GEMM (f32 out) ----------------

__global__ void __launch_bounds__(256)
proj_kernel(const unsigned short* __restrict__ Oa,
            const unsigned short* __restrict__ Wtp,
            float* __restrict__ out) {
    const int lane  = threadIdx.x & 31;
    const int wave  = threadIdx.x >> 5;
    const int half  = lane >> 4;
    const int l16   = lane & 15;
    const int mbase = blockIdx.x * 32;
    const int nbase = (blockIdx.y * 8 + wave) * 64;

    v8f acc[2][4];
    gemm32x64(Oa, Wtp, mbase, nbase, l16, half, acc);

#pragma unroll
    for (int mt = 0; mt < 2; ++mt)
#pragma unroll
        for (int ng = 0; ng < 4; ++ng)
#pragma unroll
            for (int r = 0; r < 8; ++r)
                out[(size_t)(mbase + mt * 16 + r + half * 8) * EDIM +
                    nbase + ng * 16 + l16] = acc[mt][ng][r];
}

// ---------------- host launcher ----------------

extern "C" void kernel_launch(void* const* d_in, const int* in_sizes, int n_in,
                              void* d_out, int out_size, void* d_ws, size_t ws_size,
                              hipStream_t stream) {
    const float* x  = (const float*)d_in[0];
    const float* Wq = (const float*)d_in[1];
    const float* bq = (const float*)d_in[2];
    const float* Wk = (const float*)d_in[3];
    const float* bk = (const float*)d_in[4];
    const float* Wv = (const float*)d_in[5];
    const float* bv = (const float*)d_in[6];
    const float* Wp = (const float*)d_in[7];

    char* ws = (char*)d_ws;
    unsigned short* xb  = (unsigned short*)ws; ws += (size_t)MROWS * EDIM * 2;
    unsigned short* Wtq = (unsigned short*)ws; ws += (size_t)EDIM * EDIM * 2;
    unsigned short* Wtk = (unsigned short*)ws; ws += (size_t)EDIM * EDIM * 2;
    unsigned short* Wtv = (unsigned short*)ws; ws += (size_t)EDIM * EDIM * 2;
    unsigned short* Wtp = (unsigned short*)ws; ws += (size_t)EDIM * EDIM * 2;
    unsigned short* Qr  = (unsigned short*)ws; ws += (size_t)MROWS * EDIM * 2;
    unsigned short* Kr  = (unsigned short*)ws; ws += (size_t)MROWS * EDIM * 2;
    unsigned short* Vt  = (unsigned short*)ws; ws += (size_t)MROWS * EDIM * 2;
    unsigned short* Oa  = (unsigned short*)ws; ws += (size_t)MROWS * EDIM * 2;
    float* out = (float*)d_out;

    const int nx = MROWS * EDIM;
    cvt_x_kernel<<<nx / 256, 256, 0, stream>>>(x, xb, nx);

    const int nw = EDIM * EDIM;
    cvt_wt_kernel<<<nw / 256, 256, 0, stream>>>(Wq, Wtq);
    cvt_wt_kernel<<<nw / 256, 256, 0, stream>>>(Wk, Wtk);
    cvt_wt_kernel<<<nw / 256, 256, 0, stream>>>(Wv, Wtv);
    cvt_wt_kernel<<<nw / 256, 256, 0, stream>>>(Wp, Wtp);

    qkv_rope_kernel<<<dim3(MROWS / 32, 2, 3), 256, 0, stream>>>(
        xb, Wtq, Wtk, Wtv, bq, bk, bv, Qr, Kr, Vt);

    attn_kernel<<<dim3(TSEQ / 64, BB * N_HEADS), 128, 0, stream>>>(Qr, Kr, Vt, Oa);

    proj_kernel<<<dim3(MROWS / 32, 2), 256, 0, stream>>>(Oa, Wtp, out);
}
